// DeformableConv_17566416241297
// MI455X (gfx1250) — matrix operand order repaired and checked by reading, not compiled, not run on hardware
//
#include <hip/hip_runtime.h>
#include <hip/hip_bf16.h>

// DCNv2 fused: bilinear-sample -> LDS (double-buffered) -> v_wmma_f32_16x16x32_f16.
// GEMM view: M = B*H*W pixels, N = 64 out-channels, Kred = 576 (= 9 taps * 64 ch),
// reduction reordered as j = k*64 + c so each 32-wide WMMA K-chunk is
// (one tap) x (32 input channels): tap metadata computed once, reused.
// Double-buffered A panel => single barrier per chunk, sampling overlaps WMMA.
// Bilinear accumulate via explicit FMA chains to minimize VALU on the hot path.

typedef __attribute__((ext_vector_type(16))) _Float16 v16h;
typedef __attribute__((ext_vector_type(8)))  _Float16 v8h;
typedef __attribute__((ext_vector_type(2)))  _Float16 v2h;
typedef __attribute__((ext_vector_type(8)))  float    v8f;

#define CIN   64
#define COUT  64
#define HDIM  256
#define WDIM  256
#define HW    (HDIM * WDIM)
#define KTAPS 9
#define KRED  (KTAPS * CIN)      // 576
#define BSTR  584                // padded B row stride in halves (1168 B, 16B aligned, conflict-free)
#define ASTR  40                 // padded A row stride in halves (80 B, 16B aligned, conflict-free)
#define ABUF  (64 * ASTR)        // one A-panel buffer (64 pixels x 32 halves, padded)

union FragU { v16h v; v8h h[2]; };

__global__ __launch_bounds__(256, 2)
void dcn_wmma_kernel(const float* __restrict__ x,
                     const float* __restrict__ weight,
                     const float* __restrict__ offset,
                     const float* __restrict__ mask,
                     float* __restrict__ out)
{
    __shared__ _Float16 Bsh[COUT * BSTR];   // weights as [o][j = k*64 + c], f16
    __shared__ _Float16 Ash[2 * ABUF];      // double-buffered sampled A panel

    const int t   = threadIdx.x;
    const int blk = blockIdx.x;
    const int b   = blk >> 10;              // 1024 blocks per batch image
    const int rem = blk & 1023;
    const int h   = rem >> 2;               // row
    const int w0  = (rem & 3) << 6;         // 64-pixel row segment

    // -------- stage weights (f32 -> f16) into LDS, reordered to [o][k*64+c] ----
    for (int idx = t; idx < COUT * KRED; idx += 256) {
        int o  = idx / KRED;
        int r2 = idx - o * KRED;
        int c  = r2 / KTAPS;
        int k  = r2 - c * KTAPS;
        Bsh[o * BSTR + k * CIN + c] = (_Float16)weight[idx];
    }

    // -------- per-thread sampling assignment: 4 threads per pixel ----
    const int m = t & 63;                   // pixel within segment
    const int q = t >> 6;                   // channel octet selector (0..3)
    const int w = w0 + m;

    // -------- per-wave WMMA tile assignment ----
    const int lane  = t & 31;
    const int wv    = t >> 5;               // 8 waves
    const int mtile = wv & 3;               // 16-pixel M-subtile
    const int n0    = (wv >> 2) << 5;       // 32-channel N range (0 or 32)
    const int g     = lane >> 4;            // half-wave group
    const int ln    = lane & 15;

    v8f acc0 = {};                          // out-channels n0 .. n0+15
    v8f acc1 = {};                          // out-channels n0+16 .. n0+31

    const float* xb      = x + (size_t)b * CIN * HW;
    const size_t offbase = (size_t)b * (2 * KTAPS) * HW + (size_t)h * WDIM + w;
    const size_t mbase   = (size_t)b * KTAPS * HW + (size_t)h * WDIM + w;

    // ensure weight staging is visible before first fragment loads
    __syncthreads();

    for (int k = 0; k < KTAPS; ++k) {
        // tap metadata: one load triple per pixel-thread, coalesced across lanes
        float offy = offset[offbase + (size_t)(2 * k) * HW];
        float offx = offset[offbase + (size_t)(2 * k + 1) * HW];
        float mk   = mask[mbase + (size_t)k * HW];
        if (k < KTAPS - 1) {                // gfx1250 global_prefetch_b8 for next tap
            __builtin_prefetch(&offset[offbase + (size_t)(2 * k + 2) * HW], 0, 1);
            __builtin_prefetch(&mask[mbase + (size_t)(k + 1) * HW], 0, 1);
        }

        const int ky = k / 3, kx = k - 3 * ky;
        float py  = offy + (float)(h - 1 + ky);
        float px  = offx + (float)(w - 1 + kx);
        float y0f = floorf(py), x0f = floorf(px);
        float dy  = py - y0f,   dx = px - x0f;
        int y0 = (int)y0f, x0 = (int)x0f;
        int y1 = y0 + 1,   x1 = x0 + 1;

        float vy0 = ((unsigned)y0 < (unsigned)HDIM) ? 1.f : 0.f;
        float vy1 = ((unsigned)y1 < (unsigned)HDIM) ? 1.f : 0.f;
        float vx0 = ((unsigned)x0 < (unsigned)WDIM) ? 1.f : 0.f;
        float vx1 = ((unsigned)x1 < (unsigned)WDIM) ? 1.f : 0.f;
        int y0c = min(max(y0, 0), HDIM - 1), y1c = min(max(y1, 0), HDIM - 1);
        int x0c = min(max(x0, 0), WDIM - 1), x1c = min(max(x1, 0), WDIM - 1);
        int i00 = y0c * WDIM + x0c, i01 = y0c * WDIM + x1c;
        int i10 = y1c * WDIM + x0c, i11 = y1c * WDIM + x1c;

        // mask + validity folded into bilinear weights
        float w00 = (1.f - dy) * (1.f - dx) * mk * (vy0 * vx0);
        float w01 = (1.f - dy) * dx         * mk * (vy0 * vx1);
        float w10 = dy         * (1.f - dx) * mk * (vy1 * vx0);
        float w11 = dy         * dx         * mk * (vy1 * vx1);

        #pragma unroll
        for (int half = 0; half < 2; ++half) {
            const int cc = k * 2 + half;              // global chunk index (0..17)
            _Float16* Abuf = &Ash[(cc & 1) * ABUF];   // double buffer

            // ---- sample 8 channels per thread into LDS A panel (f16) ----
            const int    cbase = half * 32 + q * 8;
            const float* xc    = xb + (size_t)cbase * HW;
            #pragma unroll
            for (int i = 0; i < 8; i += 2) {
                // FMA chains: 1 mul + 3 fma per value (v_fma_f32 co-issues with gathers)
                float v0 = __builtin_fmaf(w00, xc[i00],
                           __builtin_fmaf(w01, xc[i01],
                           __builtin_fmaf(w10, xc[i10], w11 * xc[i11])));
                xc += HW;
                float v1 = __builtin_fmaf(w00, xc[i00],
                           __builtin_fmaf(w01, xc[i01],
                           __builtin_fmaf(w10, xc[i10], w11 * xc[i11])));
                xc += HW;
                v2h p; p.x = (_Float16)v0; p.y = (_Float16)v1;   // v_cvt_pk_f16_f32
                *(v2h*)&Abuf[m * ASTR + q * 8 + i] = p;          // 4B-aligned ds_store
            }

            // single barrier per chunk: writes(buf p, chunk cc) -> reads(buf p, cc).
            // Safe vs chunk cc+2 rewrite of buf p: each wave's chunk-cc ds_loads
            // complete (s_wait_dscnt before WMMA) before it signals barrier cc+1.
            __syncthreads();

            // ---- fragment loads per CDNA5 WMMA VGPR layouts ----
            const int j0 = k * CIN + half * 32;       // chunk base in reduction dim
            FragU af, bf0, bf1;
            const _Float16* arow = &Abuf[(mtile * 16 + ln) * ASTR];
            af.h[0] = *(const v8h*)(arow + g * 8);       // K = g*8 .. g*8+7
            af.h[1] = *(const v8h*)(arow + 16 + g * 8);  // K = 16+g*8 ..
            const _Float16* br0 = &Bsh[(n0 + ln) * BSTR + j0 + g * 8];
            bf0.h[0] = *(const v8h*)(br0);               // K = g*8 .. g*8+7
            bf0.h[1] = *(const v8h*)(br0 + 16);          // K = 16+g*8 ..
            const _Float16* br1 = &Bsh[(n0 + 16 + ln) * BSTR + j0 + g * 8];
            bf1.h[0] = *(const v8h*)(br1);
            bf1.h[1] = *(const v8h*)(br1 + 16);

            acc0 = __builtin_amdgcn_wmma_f32_16x16x32_f16(
                false, af.v, false, bf0.v, (short)0, acc0, false, false);
            acc1 = __builtin_amdgcn_wmma_f32_16x16x32_f16(
                false, af.v, false, bf1.v, (short)0, acc1, false, false);
            // no second barrier: next chunk writes the other A buffer
        }
    }

    // -------- store C: VGPR r -> pixel (g*8 + r), lane -> out-channel ----
    const int wpix = w0 + mtile * 16 + g * 8;            // 32B-aligned column base
    float* op0 = out + (size_t)(b * COUT + n0 + ln) * HW + (size_t)h * WDIM + wpix;
    float* op1 = op0 + (size_t)16 * HW;
    float4 s0a = make_float4(acc0[0], acc0[1], acc0[2], acc0[3]);
    float4 s0b = make_float4(acc0[4], acc0[5], acc0[6], acc0[7]);
    float4 s1a = make_float4(acc1[0], acc1[1], acc1[2], acc1[3]);
    float4 s1b = make_float4(acc1[4], acc1[5], acc1[6], acc1[7]);
    *(float4*)(op0)     = s0a;                           // global_store_b128
    *(float4*)(op0 + 4) = s0b;
    *(float4*)(op1)     = s1a;
    *(float4*)(op1 + 4) = s1b;
}

extern "C" void kernel_launch(void* const* d_in, const int* in_sizes, int n_in,
                              void* d_out, int out_size, void* d_ws, size_t ws_size,
                              hipStream_t stream) {
    const float* x      = (const float*)d_in[0];
    const float* weight = (const float*)d_in[1];
    const float* offset = (const float*)d_in[2];
    const float* mask   = (const float*)d_in[3];
    float* out          = (float*)d_out;

    const int B = in_sizes[0] / (CIN * HW);              // = 2
    // 64-pixel segments: 4 per row * 256 rows * B
    dim3 grid((unsigned)(B * HDIM * (WDIM / 64)));
    dim3 block(256);
    dcn_wmma_kernel<<<grid, block, 0, stream>>>(x, weight, offset, mask, out);
}
